// ReDoTrAS_57861799411926
// MI455X (gfx1250) — matrix operand order repaired and checked by reading, not compiled, not run on hardware
//
#include <hip/hip_runtime.h>

// ---------------------------------------------------------------------------
// MI455X (gfx1250) implementation of the two-level stateful segment attention.
// fp32 inputs are converted to bf16; every matmul runs on
// v_wmma_f32_16x16x32_bf16 (fp32 accumulate). Weights are pre-transposed once
// to (H, N, K) so GEMM B-fragments are contiguous global loads (L2-resident,
// no LDS round trip, no barriers in the GEMM inner loop).
// ---------------------------------------------------------------------------

typedef __attribute__((ext_vector_type(16))) __bf16 v16bf;
typedef __attribute__((ext_vector_type(8)))  __bf16 v8bf;
typedef __attribute__((ext_vector_type(8)))  float  v8f;

#define DEVI __device__ __forceinline__

DEVI v16bf cat8(v8bf lo, v8bf hi) {
  v16bf o;
#pragma unroll
  for (int i = 0; i < 8; i++) { o[i] = lo[i]; o[i + 8] = hi[i]; }
  return o;
}

// A fragment (16x32 bf16). Lane l (l<16): row M=l, K {0..7},{16..23};
// lane l>=16: row M=l-16, K {8..15},{24..31}. Source row-major, leading dim lda.
DEVI v16bf load_a16(const __bf16* A, int lda, int row0, int kc, int lane) {
  int r = lane & 15, lh = lane >> 4;
  const __bf16* p = A + (size_t)(row0 + r) * lda + kc + lh * 8;
  return cat8(*(const v8bf*)p, *(const v8bf*)(p + 16));
}

// B fragment (32x16 bf16). Lane l (l<16): col N=l, K {0..7},{16..23}; lane l>=16:
// col N=l-16, K {8..15},{24..31}. Source: element (k,n) at base + n*nstride + k
// (contiguous along k for a fixed column).
DEVI v16bf load_b16(const __bf16* base, int nstride, int lane) {
  int n = lane & 15, lh = lane >> 4;
  const __bf16* p = base + (size_t)n * nstride + lh * 8;
  return cat8(*(const v8bf*)p, *(const v8bf*)(p + 16));
}

DEVI v8f wmma_bf16(v16bf a, v16bf b, v8f c) {
  return __builtin_amdgcn_wmma_f32_16x16x32_bf16(false, a, false, b,
                                                 (short)0, c, false, false);
}

// ---------------------------------------------------------------------------
// Batched per-(b,h) GEMM: Y[b,h, yrow0+m, n] = sum_k A[b,h, arow0+m, k] * W[h,k,n]
// W is pre-transposed: Wt[h, n, k] with n-stride ldwt (== K).
// Block: 128 threads (4 waves), tile 64(M) x 64(N), K-chunks of 32.
// No LDS, no barriers: A-frag + 4 B-frags are straight global b128 loads,
// then 4 back-to-back WMMAs. Optional second output row offset (dual write).
// ---------------------------------------------------------------------------
template <bool OUTF32>
__global__ __launch_bounds__(128) void gemm_bh_kernel(
    const __bf16* __restrict__ A, long sAb, long sAh, int arow0, int lda,
    const __bf16* __restrict__ Wt, long sWh, int ldwt,
    void* __restrict__ Yv, long sYb, long sYh, int yrow0, int yrow0b, int ldy,
    int H, int MT, int NT, int K) {
  int idx = blockIdx.x;
  int nt = idx % NT; idx /= NT;
  int mt = idx % MT; idx /= MT;
  int h = idx % H;
  int b = idx / H;
  int lane = threadIdx.x & 31, wave = threadIdx.x >> 5;
  const __bf16* Ab = A + (size_t)b * sAb + (size_t)h * sAh;
  const __bf16* Wh = Wt + (size_t)h * sWh;
  int m0 = mt * 64 + wave * 16;
  int n0 = nt * 64;
  v8f acc[4] = {};
  for (int kc = 0; kc < K; kc += 32) {
    v16bf a = load_a16(Ab, lda, arow0 + m0, kc, lane);
    if (kc + 32 < K)  // prefetch next A chunk (global_prefetch_b8)
      __builtin_prefetch(Ab + (size_t)(arow0 + m0 + (lane & 15)) * lda + kc + 32, 0, 3);
    v16bf b0 = load_b16(Wh + (size_t)(n0 +  0) * ldwt + kc, ldwt, lane);
    v16bf b1 = load_b16(Wh + (size_t)(n0 + 16) * ldwt + kc, ldwt, lane);
    v16bf b2 = load_b16(Wh + (size_t)(n0 + 32) * ldwt + kc, ldwt, lane);
    v16bf b3 = load_b16(Wh + (size_t)(n0 + 48) * ldwt + kc, ldwt, lane);
    acc[0] = wmma_bf16(a, b0, acc[0]);
    acc[1] = wmma_bf16(a, b1, acc[1]);
    acc[2] = wmma_bf16(a, b2, acc[2]);
    acc[3] = wmma_bf16(a, b3, acc[3]);
  }
  int lh = lane >> 4, nl = lane & 15;
#pragma unroll
  for (int t = 0; t < 4; t++)
#pragma unroll
    for (int r = 0; r < 8; r++) {
      int row = m0 + r + 8 * lh;
      int col = n0 + t * 16 + nl;
      size_t o1 = (size_t)b * sYb + (size_t)h * sYh + (size_t)(yrow0 + row) * ldy + col;
      if (OUTF32) ((float*)Yv)[o1] = acc[t][r];
      else        ((__bf16*)Yv)[o1] = (__bf16)acc[t][r];
      if (yrow0b >= 0) {
        size_t o2 = (size_t)b * sYb + (size_t)h * sYh + (size_t)(yrow0b + row) * ldy + col;
        if (OUTF32) ((float*)Yv)[o2] = acc[t][r];
        else        ((__bf16*)Yv)[o2] = (__bf16)acc[t][r];
      }
    }
}

// ---------------------------------------------------------------------------
// Causal flash attention over L=768, head dim E (64 or 128).
// Grid: (B*H*6), block 256 (8 waves); each wave owns 16 query rows; key tiles
// of 32. Scores via WMMA (K^T fragments are contiguous global loads), online
// softmax with shfl_xor row-reductions over the 16-lane halves, P transposed
// C-layout -> A-layout through wave-private LDS, V staged transposed in LDS.
// ---------------------------------------------------------------------------
template <int E>
__global__ __launch_bounds__(256) void attn_kernel(
    const __bf16* __restrict__ Q, const __bf16* __restrict__ K,
    const __bf16* __restrict__ V, __bf16* __restrict__ O) {
  constexpr int L = 768;
  __shared__ __bf16 VT[E * 32];       // [e][key_local]
  __shared__ __bf16 P[8 * 16 * 32];   // per-wave 16x32 P tile
  int tid = threadIdx.x, lane = tid & 31, wave = tid >> 5;
  int rb = blockIdx.x % 6, bh = blockIdx.x / 6;
  const __bf16* Qb = Q + (size_t)bh * L * E;
  const __bf16* Kb = K + (size_t)bh * L * E;
  const __bf16* Vb = V + (size_t)bh * L * E;
  __bf16* Ob = O + (size_t)bh * L * E;
  int q0 = rb * 128 + wave * 16;
  int lh = lane >> 4, nl = lane & 15;
  v16bf qf[E / 32];
#pragma unroll
  for (int ec = 0; ec < E; ec += 32) qf[ec / 32] = load_a16(Qb, E, q0, ec, lane);
  float m8[8], s8[8];
#pragma unroll
  for (int r = 0; r < 8; r++) { m8[r] = -3.0e38f; s8[r] = 0.f; }
  v8f oacc[E / 16] = {};
  __bf16* Pw = &P[wave * 16 * 32];
  int kend = rb * 128 + 128;
  for (int kb = 0; kb < kend; kb += 32) {
    __syncthreads();
    // stage V[kb..kb+32, :] transposed into LDS
#pragma unroll
    for (int j = tid; j < (32 * E / 8); j += 256) {
      int e = j * 8, key = e / E, e0 = e % E;
      v8bf v = *(const v8bf*)(Vb + (size_t)(kb + key) * E + e0);
#pragma unroll
      for (int i = 0; i < 8; i++) VT[(e0 + i) * 32 + key] = v[i];
    }
    __syncthreads();
    // score tiles: keys [kb,kb+16) and [kb+16,kb+32)
    v8f sc0 = {}, sc1 = {};
#pragma unroll
    for (int ec = 0; ec < E; ec += 32) {
      v16bf bk0 = load_b16(Kb + (size_t)kb * E + ec, E, lane);
      v16bf bk1 = load_b16(Kb + (size_t)(kb + 16) * E + ec, E, lane);
      sc0 = wmma_bf16(qf[ec / 32], bk0, sc0);
      sc1 = wmma_bf16(qf[ec / 32], bk1, sc1);
    }
    float f[8];
#pragma unroll
    for (int r = 0; r < 8; r++) {
      int row = q0 + r + 8 * lh;
      float x0 = (kb + nl      <= row) ? sc0[r] * 0.125f : -3.0e38f;
      float x1 = (kb + 16 + nl <= row) ? sc1[r] * 0.125f : -3.0e38f;
      float mx = fmaxf(x0, x1);
#pragma unroll
      for (int d = 1; d < 16; d <<= 1) mx = fmaxf(mx, __shfl_xor(mx, d, 16));
      float mn = fmaxf(m8[r], mx);
      f[r] = __expf(m8[r] - mn);
      m8[r] = mn;
      float p0 = __expf(x0 - mn);
      float p1 = __expf(x1 - mn);
      float rs = p0 + p1;
#pragma unroll
      for (int d = 1; d < 16; d <<= 1) rs += __shfl_xor(rs, d, 16);
      s8[r] = s8[r] * f[r] + rs;
      Pw[(r + 8 * lh) * 32 + nl]      = (__bf16)p0;
      Pw[(r + 8 * lh) * 32 + 16 + nl] = (__bf16)p1;
    }
#pragma unroll
    for (int t = 0; t < E / 16; t++)
#pragma unroll
      for (int r = 0; r < 8; r++) oacc[t][r] *= f[r];
    v16bf pf = load_a16(Pw, 32, 0, 0, lane);  // wave-private: DS ops in-order
#pragma unroll
    for (int t = 0; t < E / 16; t += 2) {     // paired loads, paired WMMAs
      v16bf bv0 = load_b16(&VT[(t + 0) * 16 * 32], 32, lane);
      v16bf bv1 = load_b16(&VT[(t + 1) * 16 * 32], 32, lane);
      oacc[t + 0] = wmma_bf16(pf, bv0, oacc[t + 0]);
      oacc[t + 1] = wmma_bf16(pf, bv1, oacc[t + 1]);
    }
  }
  float inv[8];
#pragma unroll
  for (int r = 0; r < 8; r++) inv[r] = (s8[r] > 0.f) ? 1.f / s8[r] : 0.f;
#pragma unroll
  for (int t = 0; t < E / 16; t++)
#pragma unroll
    for (int r = 0; r < 8; r++) {
      int row = q0 + r + 8 * lh, col = t * 16 + nl;
      Ob[(size_t)row * E + col] = (__bf16)(oacc[t][r] * inv[r]);
    }
}

// fp32 -> bf16 elementwise convert
__global__ void cvt_kernel(const float* __restrict__ s, __bf16* __restrict__ d, long n) {
  long i = ((long)blockIdx.x * blockDim.x + threadIdx.x) * 4;
#pragma unroll
  for (int u = 0; u < 4; u++)
    if (i + u < n) d[i + u] = (__bf16)s[i + u];
}

// fp32 (H,K,N) -> bf16 transposed (H,N,K): one-time weight relayout so GEMM
// B-fragments are k-contiguous global loads.
__global__ void cvtT_kernel(const float* __restrict__ s, __bf16* __restrict__ d,
                            int Hh, int K, int N) {
  long total = (long)Hh * K * N;
  long i = (long)blockIdx.x * blockDim.x + threadIdx.x;
  if (i >= total) return;
  int n = (int)(i % N);
  long t = i / N;
  int k = (int)(t % K);
  int h = (int)(t / K);
  d[((size_t)h * N + n) * K + k] = (__bf16)s[i];
}

// Flatten att2 slice rows [r0, r0+rows) of each (b,h) into (B, H*rows*128)
// contiguous (this IS the reference's reshape, which flattens h-major).
__global__ void repack_kernel(const __bf16* __restrict__ src, __bf16* __restrict__ dst,
                              int rows, int r0) {
  const long per = (long)rows * 128;
  const long total = 32 * per;  // B*H = 32
  long i = ((long)blockIdx.x * blockDim.x + threadIdx.x) * 8;
  if (i >= total) return;
  int bh = (int)(i / per);
  long rem = i % per;
  int b = bh >> 4, h = bh & 15;
  v8bf v = *(const v8bf*)(src + (size_t)bh * 768 * 128 + (size_t)r0 * 128 + rem);
  *(v8bf*)(dst + (size_t)b * 16 * per + (size_t)h * per + rem) = v;
}

__global__ void copyf_kernel(const float* __restrict__ s, float* __restrict__ d, long n) {
  long i = (long)blockIdx.x * blockDim.x + threadIdx.x;
  if (i < n) d[i] = s[i];
}

extern "C" void kernel_launch(void* const* d_in, const int* in_sizes, int n_in,
                              void* d_out, int out_size, void* d_ws, size_t ws_size,
                              hipStream_t stream) {
  (void)in_sizes; (void)n_in; (void)out_size; (void)ws_size;
  constexpr int B = 2, S = 2048, D = 1024, H = 16, DK = 64, DV = 64;
  constexpr int SEG = 512, SL = 128, L = 768;

  const float* x = (const float*)d_in[0];
  const float* state = (const float*)d_in[1];

  // ---- workspace bump allocation ----
  size_t off = 0;
  auto alloc = [&](size_t bytes) -> char* {
    char* p = (char*)d_ws + off;
    off += (bytes + 255) & ~(size_t)255;
    return p;
  };
  __bf16* xbf = (__bf16*)alloc((size_t)B * S * D * 2);
  // per-weight (Hh, K, N) for the transposed relayout
  int wH[17] = {0, 0, H, H, H, H, 1, H, H, H, H, H, H, H, H, H, 1};
  int wK[17] = {0, 0, D, D, D, DV, 2 * H * DV, D, D, D, DV, DV, DV, DV, DV, DV, 2 * H * DV};
  int wN[17] = {0, 0, DK, DK, DV, 2 * DK, D, DK, DK, DV, 2 * DK, 2 * DK, 2 * DV, 2 * DK, 2 * DK, 2 * DV, D};
  __bf16* wbf[17];
  for (int i = 2; i < 17; i++)
    wbf[i] = (__bf16*)alloc((size_t)wH[i] * wK[i] * wN[i] * 2);
  __bf16* stbf = (__bf16*)alloc((size_t)B * SL * D * 2);
  float*  stf  = (float*)alloc((size_t)B * SL * D * 4);
  __bf16* kk  = (__bf16*)alloc((size_t)B * H * L * DK * 2);
  __bf16* qq  = (__bf16*)alloc((size_t)B * H * L * DK * 2);
  __bf16* vv  = (__bf16*)alloc((size_t)B * H * L * DV * 2);
  __bf16* att = (__bf16*)alloc((size_t)B * H * L * DV * 2);
  __bf16* k2  = (__bf16*)alloc((size_t)B * H * L * 128 * 2);
  __bf16* q2  = (__bf16*)alloc((size_t)B * H * L * 128 * 2);
  __bf16* v2  = (__bf16*)alloc((size_t)B * H * L * 128 * 2);
  __bf16* at2 = (__bf16*)alloc((size_t)B * H * L * 128 * 2);
  __bf16* a2c = (__bf16*)alloc((size_t)B * SEG * 2048 * 2);
  __bf16* a2e = (__bf16*)alloc((size_t)B * SL * 2048 * 2);

  auto cvt = [&](const float* s, __bf16* dd, long n) {
    int blocks = (int)((n + 1023) / 1024);
    cvt_kernel<<<blocks, 256, 0, stream>>>(s, dd, n);
  };

  // ---- one-time conversions ----
  cvt(x, xbf, (long)B * S * D);
  for (int i = 2; i < 17; i++) {
    long n = (long)wH[i] * wK[i] * wN[i];
    int blocks = (int)((n + 255) / 256);
    cvtT_kernel<<<blocks, 256, 0, stream>>>((const float*)d_in[i], wbf[i],
                                            wH[i], wK[i], wN[i]);
  }
  cvt(state, stbf, (long)SL * D);                   // broadcast to b=0
  cvt(state, stbf + (size_t)SL * D, (long)SL * D);  // and b=1

  auto gemm = [&](const __bf16* A, long sAb, long sAh, int arow0, int lda,
                  const __bf16* Wt, long sWh, int ldwt,
                  void* Y, long sYb, long sYh, int yrow0, int yrow0b, int ldy,
                  int Hh, int M, int N, int K, bool f32out) {
    int MT = M / 64, NT = N / 64;
    dim3 g((unsigned)(B * Hh * MT * NT));
    if (f32out)
      gemm_bh_kernel<true><<<g, 128, 0, stream>>>(A, sAb, sAh, arow0, lda, Wt, sWh, ldwt,
                                                  Y, sYb, sYh, yrow0, yrow0b, ldy, Hh, MT, NT, K);
    else
      gemm_bh_kernel<false><<<g, 128, 0, stream>>>(A, sAb, sAh, arow0, lda, Wt, sWh, ldwt,
                                                   Y, sYb, sYh, yrow0, yrow0b, ldy, Hh, MT, NT, K);
  };

  const long sKb = (long)H * L * DK, sKh = (long)L * DK;
  const long s2b = (long)H * L * 128, s2h = (long)L * 128;
  const long swD64 = (long)D * DK;     // per-head elems of (64,1024)-T weights
  const long sw64 = (long)DV * 128;    // per-head elems of (128,64)-T weights

  for (int t = 0; t < 4; t++) {
    // segment projections -> rows [SL, SL+SEG)
    gemm(xbf, (long)S * D, 0, t * SEG, D, wbf[2], swD64, D, kk, sKb, sKh, SL, -1, DK, H, SEG, DK, D, false);
    gemm(xbf, (long)S * D, 0, t * SEG, D, wbf[3], swD64, D, qq, sKb, sKh, SL, -1, DK, H, SEG, DK, D, false);
    gemm(xbf, (long)S * D, 0, t * SEG, D, wbf[4], swD64, D, vv, sKb, sKh, SL, -1, DV, H, SEG, DV, D, false);
    // state projections -> rows [0,SL) and [L-SL,L) (dual write)
    gemm(stbf, (long)SL * D, 0, 0, D, wbf[7], swD64, D, kk, sKb, sKh, 0, L - SL, DK, H, SL, DK, D, false);
    gemm(stbf, (long)SL * D, 0, 0, D, wbf[8], swD64, D, qq, sKb, sKh, 0, L - SL, DK, H, SL, DK, D, false);
    gemm(stbf, (long)SL * D, 0, 0, D, wbf[9], swD64, D, vv, sKb, sKh, 0, L - SL, DV, H, SL, DV, D, false);
    // attention level 1 (E=64)
    attn_kernel<64><<<dim3(B * H * 6), 256, 0, stream>>>(qq, kk, vv, att);
    // level-2 projections (middle uses W2k for k2/q2/v2, as in the reference)
    gemm(att, sKb, sKh, SL, DK, wbf[5],  sw64, DV, k2, s2b, s2h, SL, -1, 128, H, SEG, 128, DV, false);
    gemm(att, sKb, sKh, SL, DK, wbf[5],  sw64, DV, q2, s2b, s2h, SL, -1, 128, H, SEG, 128, DV, false);
    gemm(att, sKb, sKh, SL, DK, wbf[5],  sw64, DV, v2, s2b, s2h, SL, -1, 128, H, SEG, 128, DV, false);
    gemm(att, sKb, sKh, 0,  DK, wbf[10], sw64, DV, k2, s2b, s2h, 0,  -1, 128, H, SL, 128, DV, false);
    gemm(att, sKb, sKh, 0,  DK, wbf[11], sw64, DV, q2, s2b, s2h, 0,  -1, 128, H, SL, 128, DV, false);
    gemm(att, sKb, sKh, 0,  DK, wbf[12], sw64, DV, v2, s2b, s2h, 0,  -1, 128, H, SL, 128, DV, false);
    gemm(att, sKb, sKh, L - SL, DK, wbf[13], sw64, DV, k2, s2b, s2h, L - SL, -1, 128, H, SL, 128, DV, false);
    gemm(att, sKb, sKh, L - SL, DK, wbf[14], sw64, DV, q2, s2b, s2h, L - SL, -1, 128, H, SL, 128, DV, false);
    gemm(att, sKb, sKh, L - SL, DK, wbf[15], sw64, DV, v2, s2b, s2h, L - SL, -1, 128, H, SL, 128, DV, false);
    // attention level 2 (E=128)
    attn_kernel<128><<<dim3(B * H * 6), 256, 0, stream>>>(q2, k2, v2, at2);
    // repack middle / end slices into (B, 512|128, 2048)
    repack_kernel<<<1024, 256, 0, stream>>>(at2, a2c, SEG, SL);
    repack_kernel<<<256, 256, 0, stream>>>(at2, a2e, SL, L - SL);
    // output projection: out[b, t*SEG + m, :] = a2c @ Wout  (fp32)
    gemm(a2c, (long)SEG * 2048, 0, 0, 2048, wbf[6], 0, 2 * H * DV,
         (float*)d_out, (long)S * D, 0, t * SEG, -1, D, 1, SEG, D, 2048, true);
    // new state: stf = a2e @ Wout_state (fp32), then bf16 for next step
    gemm(a2e, (long)SL * 2048, 0, 0, 2048, wbf[16], 0, 2 * H * DV,
         stf, (long)SL * D, 0, 0, -1, D, 1, SL, D, 2048, true);
    cvt(stf, stbf, (long)B * SL * D);
  }
  // final state -> tail of d_out
  copyf_kernel<<<(B * SL * D + 255) / 256, 256, 0, stream>>>(
      stf, (float*)d_out + (size_t)B * S * D, (long)B * SL * D);
}
